// DPSA3D_30021821399893
// MI455X (gfx1250) — matrix operand order repaired and checked by reading, not compiled, not run on hardware
//
#include <hip/hip_runtime.h>
#include <hip/hip_bf16.h>
#include <math.h>

// ---------- WMMA types & fragment helpers (gfx1250, wave32) ----------
typedef __attribute__((ext_vector_type(16))) _Float16 v16h;
typedef __attribute__((ext_vector_type(8)))  _Float16 h8;
typedef __attribute__((ext_vector_type(8)))  float    v8f;
typedef __attribute__((ext_vector_type(4)))  float    f4;

// A fragment 16x32 f16 from row-major [row][k] LDS, stride in halfs (stride*2 % 16 == 0).
// lanes 0-15: row=lane, K={half*8+0..7, 16+half*8+0..7} -> two contiguous b128 runs.
__device__ inline v16h load_a_frag(const _Float16* lds, int row0, int k0, int stride, int lane) {
  const _Float16* p = lds + (size_t)(row0 + (lane & 15)) * stride + k0 + ((lane >> 4) * 8);
  h8 lo = *(const h8*)p;          // e0..7   : K = base .. base+7
  h8 hi = *(const h8*)(p + 16);   // e8..15  : K = base+16 .. base+23
  return __builtin_shufflevector(lo, hi, 0,1,2,3,4,5,6,7,8,9,10,11,12,13,14,15);
}

// B fragment 32x16 f16 from an [n][k] LDS layout: K={half*8+0..7, 16+half*8+0..7} -> two contiguous b128 runs.
__device__ inline v16h load_b_fragT(const _Float16* lds, int k0, int n0, int stride, int lane) {
  const _Float16* p = lds + (size_t)(n0 + (lane & 15)) * stride + k0 + ((lane >> 4) * 8);
  h8 lo = *(const h8*)p;
  h8 hi = *(const h8*)(p + 16);
  return __builtin_shufflevector(lo, hi, 0,1,2,3,4,5,6,7,8,9,10,11,12,13,14,15);
}

// C/D f32 16x16: VGPR r -> row r (lanes 0-15) / row r+8 (lanes 16-31), N = lane&15
__device__ inline void store_c_lds(float* lds, const v8f& c, int row0, int col0, int stride, int lane) {
  int n = col0 + (lane & 15);
  int m = row0 + ((lane >> 4) ? 8 : 0);
#pragma unroll
  for (int r = 0; r < 8; ++r) lds[(size_t)(m + r) * stride + n] = c[r];
}

// ---------- problem constants ----------
#define NVOX   65536
#define SPAT   32768
#define CIN    128
#define INNER  512
#define OC3    1536
#define DH     64
#define BH     16
#define LPAD   128

// ---------- K0: convert weights to f16 ----------
__global__ void k_cvt(const float* __restrict__ wq, const float* __restrict__ wo,
                      _Float16* wqh, _Float16* woh) {
  int i = blockIdx.x * 256 + threadIdx.x;
  if (i < OC3 * CIN) wqh[i] = (_Float16)wq[i];
  if (i < DH * INNER) woh[i] = (_Float16)wo[i];
}

// ---------- K1: LayerNorm over C, transpose to voxel-major f16 ----------
__global__ void k_layernorm(const float* __restrict__ x, const float* __restrict__ g,
                            const float* __restrict__ be, _Float16* __restrict__ xn) {
  int v = blockIdx.x * 256 + threadIdx.x;
  int b = v >> 15, s = v & (SPAT - 1);
  const float* xp = x + (size_t)b * CIN * SPAT + s;
  float sum = 0.f, sum2 = 0.f;
  for (int c = 0; c < CIN; ++c) { float t = xp[(size_t)c * SPAT]; sum += t; sum2 += t * t; }
  float mean = sum * (1.f / CIN);
  float var  = sum2 * (1.f / CIN) - mean * mean;
  float inv  = 1.f / (sqrtf(fmaxf(var, 0.f)) + 1e-6f);
  _Float16* op = xn + (size_t)v * CIN;
  for (int c8 = 0; c8 < CIN; c8 += 8) {
    h8 hv;
#pragma unroll
    for (int e = 0; e < 8; ++e) {
      float t = xp[(size_t)(c8 + e) * SPAT];
      hv[e] = (_Float16)(g[c8 + e] * (t - mean) * inv + be[c8 + e]);
    }
    *(h8*)(op + c8) = hv;
  }
}

// ---------- K2: QKV GEMM (WMMA) + fused per-head L2 norm ----------
// grid (24 col tiles, 1024 row tiles), block 256. Full 64x128 A and B panels staged once.
__global__ __launch_bounds__(256) void k_qkv(const _Float16* __restrict__ xn,
                                             const _Float16* __restrict__ wq,
                                             _Float16* qn, _Float16* kn, _Float16* vn) {
  __shared__ alignas(16) _Float16 aF[64 * 136];
  __shared__ alignas(16) _Float16 bF[64 * 136];
  __shared__ alignas(16) float    acc[64][64];
  __shared__ float    rscale[64];
  int ct = blockIdx.x, rt = blockIdx.y;
  int tid = threadIdx.x, wave = tid >> 5, lane = tid & 31;
  int mi = wave & 3, nb = (wave >> 2) * 2;
  // stage panels (128-bit copies)
  for (int i = tid; i < 1024; i += 256) {
    int r = i >> 4, c8 = (i & 15) * 8;
    *(h8*)(aF + r * 136 + c8) = *(const h8*)(xn + (size_t)(rt * 64 + r) * CIN + c8);
    *(h8*)(bF + r * 136 + c8) = *(const h8*)(wq + (size_t)(ct * 64 + r) * CIN + c8);
  }
  __syncthreads();
  v8f c0 = {}, c1 = {};
#pragma unroll
  for (int k0 = 0; k0 < CIN; k0 += 32) {
    v16h a  = load_a_frag(aF, mi * 16, k0, 136, lane);
    v16h b0 = load_b_fragT(bF, k0, nb * 16, 136, lane);
    v16h b1 = load_b_fragT(bF, k0, (nb + 1) * 16, 136, lane);
    c0 = __builtin_amdgcn_wmma_f32_16x16x32_f16(false, a, false, b0, (short)0, c0, false, false);
    c1 = __builtin_amdgcn_wmma_f32_16x16x32_f16(false, a, false, b1, (short)0, c1, false, false);
  }
  __syncthreads();
  store_c_lds(&acc[0][0], c0, mi * 16, nb * 16, 64, lane);
  store_c_lds(&acc[0][0], c1, mi * 16, (nb + 1) * 16, 64, lane);
  __syncthreads();
  bool isV = (ct >= 16);
  if (!isV) {  // 64-col tile == one head: L2 norm over cols per row
    int r = tid >> 2, part = tid & 3;
    float ss = 0.f;
    for (int n = part; n < 64; n += 4) { float t = acc[r][n]; ss += t * t; }
    ss += __shfl_xor(ss, 1, 32);
    ss += __shfl_xor(ss, 2, 32);
    if (part == 0) rscale[r] = 1.f / (sqrtf(ss) + 1e-6f);
    __syncthreads();
  }
  int hd = ct & 7;
  _Float16* dst = isV ? vn : (ct < 8 ? qn : kn);
  for (int i = tid; i < 512; i += 256) {
    int r = i >> 3, c8 = (i & 7) * 8;
    int vr = rt * 64 + r;
    int b = vr >> 15, s = vr & (SPAT - 1);
    float sc = isV ? 1.f : rscale[r];
    h8 hv;
#pragma unroll
    for (int e = 0; e < 8; ++e) hv[e] = (_Float16)(acc[r][c8 + e] * sc);
    *(h8*)(dst + ((size_t)(b * 8 + hd) * SPAT + s) * DH + c8) = hv;
  }
}

// ---------- K3: |q| probe + |k| axis sums ----------
__global__ __launch_bounds__(256) void k_reduce(const _Float16* __restrict__ qn,
                                                const _Float16* __restrict__ kn,
                                                float* probe, float* ksd, float* ksh, float* ksw) {
  __shared__ float sp[64];
  __shared__ float sd[64][32];
  __shared__ float shh[64][32];
  __shared__ float sw[64][32];
  int bh = blockIdx.x, tid = threadIdx.x;
  for (int i = tid; i < 64; i += 256) sp[i] = 0.f;
  for (int i = tid; i < 2048; i += 256) { (&sd[0][0])[i] = 0.f; (&shh[0][0])[i] = 0.f; (&sw[0][0])[i] = 0.f; }
  __syncthreads();
  int c8 = (tid & 7) * 8;
  int so = tid >> 3;                 // this thread's fixed w coordinate (0..31)
  const _Float16* qp = qn + (size_t)bh * SPAT * DH;
  const _Float16* kp = kn + (size_t)bh * SPAT * DH;
  float p[8] = {}, kw8[8] = {};
  for (int d = 0; d < 32; ++d) {
    float kd8[8] = {};
    for (int i = 0; i < 32; ++i) {
      int hh = (i + so) & 31;        // stagger to avoid LDS atomic collisions
      int s = (d << 10) + (hh << 5) + so;
      h8 q8 = *(const h8*)(qp + (size_t)s * DH + c8);
      h8 k8 = *(const h8*)(kp + (size_t)s * DH + c8);
#pragma unroll
      for (int e = 0; e < 8; ++e) {
        float qv = fabsf((float)q8[e]);
        float kv = fabsf((float)k8[e]);
        p[e] += qv; kd8[e] += kv; kw8[e] += kv;
        atomicAdd(&shh[c8 + e][hh], kv);
      }
    }
#pragma unroll
    for (int e = 0; e < 8; ++e) atomicAdd(&sd[c8 + e][d], kd8[e]);
  }
#pragma unroll
  for (int e = 0; e < 8; ++e) { atomicAdd(&sp[c8 + e], p[e]); atomicAdd(&sw[c8 + e][so], kw8[e]); }
  __syncthreads();
  for (int i = tid; i < 64; i += 256) probe[bh * 64 + i] = sp[i];
  for (int i = tid; i < 2048; i += 256) {
    ksd[bh * 2048 + i] = (&sd[0][0])[i];
    ksh[bh * 2048 + i] = (&shh[0][0])[i];
    ksw[bh * 2048 + i] = (&sw[0][0])[i];
  }
}

// ---------- K4: scores + top-5 per axis ----------
__global__ void k_topk(const float* __restrict__ probe, const float* __restrict__ ksd,
                       const float* __restrict__ ksh, const float* __restrict__ ksw, int* idx) {
  int bh = blockIdx.x, lane = threadIdx.x;   // 32 threads
  __shared__ float sc[3][32];
  const float* pr = probe + bh * 64;
  const float* arrs[3] = { ksd + bh * 2048, ksh + bh * 2048, ksw + bh * 2048 };
  for (int a = 0; a < 3; ++a) {
    float s = 0.f;
    for (int ch = 0; ch < 64; ++ch) s += pr[ch] * arrs[a][ch * 32 + lane];
    sc[a][lane] = s;
  }
  __syncthreads();
  if (lane == 0) {
    for (int a = 0; a < 3; ++a)
      for (int t = 0; t < 5; ++t) {
        int best = 0; float bv = -1e30f;
        for (int i = 0; i < 32; ++i) if (sc[a][i] > bv) { bv = sc[a][i]; best = i; }
        idx[(bh * 3 + a) * 5 + t] = best;
        sc[a][best] = -1e30f;
      }
  }
}

// ---------- K5: gather 125 (pad 128) k (row-major) and v (transposed [ch][j]) ----------
__global__ void k_gather(const _Float16* __restrict__ kn, const _Float16* __restrict__ vn,
                         const int* __restrict__ idx, _Float16* kf, _Float16* vfT) {
  int bh = blockIdx.x;
  const int* id = idx + bh * 15;
  __shared__ int ss[125];
  for (int j = threadIdx.x; j < 125; j += 256) {
    int a = j / 25, bq = (j / 5) % 5, cq = j % 5;
    ss[j] = (id[a] * 32 + id[5 + bq]) * 32 + id[10 + cq];
  }
  __syncthreads();
  for (int i = threadIdx.x; i < LPAD * DH; i += 256) {      // kf[j][ch]
    int j = i >> 6, ch = i & 63;
    _Float16 kv = (_Float16)0.f;
    if (j < 125) kv = kn[((size_t)bh * SPAT + ss[j]) * DH + ch];
    kf[(size_t)bh * LPAD * DH + i] = kv;
  }
  for (int i = threadIdx.x; i < DH * LPAD; i += 256) {      // vfT[ch][j]
    int ch = i >> 7, j = i & 127;
    _Float16 vv = (_Float16)0.f;
    if (j < 125) vv = vn[((size_t)bh * SPAT + ss[j]) * DH + ch];
    vfT[(size_t)bh * DH * LPAD + i] = vv;
  }
}

// ---------- K6: fused attention: QK^T (WMMA) -> softmax -> attn*V (WMMA) ----------
// grid (512 row tiles, 16 bh), block 256. attn_out aliases qn (block reads its q rows first).
__global__ __launch_bounds__(256) void k_attn(const _Float16* __restrict__ qn,
                                              const _Float16* __restrict__ kf,
                                              const _Float16* __restrict__ vfT,
                                              _Float16* attn_out) {
  __shared__ alignas(16) _Float16 qT[64 * 72];
  __shared__ alignas(16) _Float16 kvT[128 * 72];   // kf as [j][72]; later vT as [ch][136]
  __shared__ alignas(16) float    sim[64 * 128];
  _Float16* simH = (_Float16*)sim;                 // f16 probs, stride 136
  int rt = blockIdx.x, bh = blockIdx.y;
  int tid = threadIdx.x, wave = tid >> 5, lane = tid & 31;
  const _Float16* qp  = qn  + ((size_t)bh * SPAT + rt * 64) * DH;
  const _Float16* kfp = kf  + (size_t)bh * LPAD * DH;
  const _Float16* vtp = vfT + (size_t)bh * DH * LPAD;
  for (int i = tid; i < 512; i += 256) {           // q: 64 rows x 8 chunks
    int r = i >> 3, c8 = (i & 7) * 8;
    *(h8*)(qT + r * 72 + c8) = *(const h8*)(qp + r * 64 + c8);
  }
  for (int i = tid; i < 1024; i += 256) {          // kf: 128 rows x 8 chunks
    int j = i >> 3, c8 = (i & 7) * 8;
    *(h8*)(kvT + j * 72 + c8) = *(const h8*)(kfp + j * 64 + c8);
  }
  __syncthreads();
  // sim = q * kf^T : M=64, N=128, K=64
  int mi = wave & 3;
  int nb0 = (wave >> 2) * 4;
  v8f cs[4] = { {}, {}, {}, {} };
#pragma unroll
  for (int k0 = 0; k0 < DH; k0 += 32) {
    v16h a = load_a_frag(qT, mi * 16, k0, 72, lane);
#pragma unroll
    for (int t = 0; t < 4; ++t) {
      v16h b = load_b_fragT(kvT, k0, (nb0 + t) * 16, 72, lane);
      cs[t] = __builtin_amdgcn_wmma_f32_16x16x32_f16(false, a, false, b, (short)0, cs[t], false, false);
    }
  }
#pragma unroll
  for (int t = 0; t < 4; ++t) store_c_lds(sim, cs[t], mi * 16, (nb0 + t) * 16, 128, lane);
  __syncthreads();
  // overlay V^T into kvT (kf consumed): [ch][136]
  for (int i = tid; i < 1024; i += 256) {
    int ch = i >> 4, c8 = (i & 15) * 8;
    *(h8*)(kvT + ch * 136 + c8) = *(const h8*)(vtp + ch * 128 + c8);
  }
  // softmax over 125 (cols >=125 masked); 4 lanes per row; regs first, barrier, write f16 probs
  int row = tid >> 2, part = tid & 3;
  float vals[32];
  const float* srow = sim + row * 128 + part * 32;
#pragma unroll
  for (int i = 0; i < 8; ++i) {
    f4 t = *(const f4*)(srow + i * 4);
#pragma unroll
    for (int e = 0; e < 4; ++e) vals[i * 4 + e] = t[e];
  }
  if (part == 3) { vals[29] = -1e30f; vals[30] = -1e30f; vals[31] = -1e30f; }
  float mx = -1e30f;
#pragma unroll
  for (int i = 0; i < 32; ++i) mx = fmaxf(mx, vals[i]);
  mx = fmaxf(mx, __shfl_xor(mx, 1, 32));
  mx = fmaxf(mx, __shfl_xor(mx, 2, 32));
  float sum = 0.f;
#pragma unroll
  for (int i = 0; i < 32; ++i) { float e = __expf(vals[i] - mx); vals[i] = e; sum += e; }
  sum += __shfl_xor(sum, 1, 32);
  sum += __shfl_xor(sum, 2, 32);
  float rs = 1.f / sum;
  __syncthreads();               // all sim reads (and V^T staging) complete
  _Float16* prow = simH + row * 136 + part * 32;
#pragma unroll
  for (int i = 0; i < 4; ++i) {
    h8 hv;
#pragma unroll
    for (int e = 0; e < 8; ++e) hv[e] = (_Float16)(vals[i * 8 + e] * rs);
    *(h8*)(prow + i * 8) = hv;
  }
  __syncthreads();
  // out = probs * V : M=64, N=64, K=128 (A stride 136, B = V^T [n][k] stride 136)
  int nb = (wave >> 2) * 2;
  v8f o0 = {}, o1 = {};
#pragma unroll
  for (int k0 = 0; k0 < LPAD; k0 += 32) {
    v16h a  = load_a_frag(simH, mi * 16, k0, 136, lane);
    v16h b0 = load_b_fragT(kvT, k0, nb * 16, 136, lane);
    v16h b1 = load_b_fragT(kvT, k0, (nb + 1) * 16, 136, lane);
    o0 = __builtin_amdgcn_wmma_f32_16x16x32_f16(false, a, false, b0, (short)0, o0, false, false);
    o1 = __builtin_amdgcn_wmma_f32_16x16x32_f16(false, a, false, b1, (short)0, o1, false, false);
  }
  __syncthreads();               // everyone done reading probs before sim is reused as f32 acc
  store_c_lds(sim, o0, mi * 16, nb * 16, 64, lane);
  store_c_lds(sim, o1, mi * 16, (nb + 1) * 16, 64, lane);
  __syncthreads();
  _Float16* op = attn_out + ((size_t)bh * SPAT + rt * 64) * DH;
  for (int i = tid; i < 512; i += 256) {
    int r = i >> 3, c8 = (i & 7) * 8;
    h8 hv;
#pragma unroll
    for (int e = 0; e < 8; ++e) hv[e] = (_Float16)sim[r * 64 + c8 + e];
    *(h8*)(op + r * 64 + c8) = hv;
  }
}

// ---------- K7: output projection (WMMA) + bias, transpose to (b,o,s) ----------
__global__ __launch_bounds__(256) void k_out(const _Float16* __restrict__ attn_out,
                                             const _Float16* __restrict__ wo,
                                             const float* __restrict__ b_out, float* out) {
  __shared__ alignas(16) _Float16 aT[64 * 48];
  __shared__ alignas(16) _Float16 bT[64 * 48];
  int rt = blockIdx.x;
  int tid = threadIdx.x, wave = tid >> 5, lane = tid & 31;
  int mi = wave & 3, nb = (wave >> 2) * 2;
  v8f c0 = {}, c1 = {};
  for (int k0 = 0; k0 < INNER; k0 += 32) {
    {
      int r = tid >> 2, c8 = (tid & 3) * 8;
      int vr = rt * 64 + r;
      int b = vr >> 15, s = vr & (SPAT - 1);
      int c = k0 + c8;  // head = c>>6 constant across the 8-half chunk
      *(h8*)(aT + r * 48 + c8) =
          *(const h8*)(attn_out + ((size_t)(b * 8 + (c >> 6)) * SPAT + s) * DH + (c & 63));
      *(h8*)(bT + r * 48 + c8) = *(const h8*)(wo + (size_t)r * INNER + k0 + c8);
    }
    __syncthreads();
    v16h a  = load_a_frag(aT, mi * 16, 0, 48, lane);
    v16h b0 = load_b_fragT(bT, 0, nb * 16, 48, lane);
    v16h b1 = load_b_fragT(bT, 0, (nb + 1) * 16, 48, lane);
    c0 = __builtin_amdgcn_wmma_f32_16x16x32_f16(false, a, false, b0, (short)0, c0, false, false);
    c1 = __builtin_amdgcn_wmma_f32_16x16x32_f16(false, a, false, b1, (short)0, c1, false, false);
    __syncthreads();
  }
  int mrow = mi * 16 + ((lane >> 4) ? 8 : 0);
  int col = lane & 15;
  int vr0 = rt * 64 + mrow;
  int b = vr0 >> 15, s0 = vr0 & (SPAT - 1);
  int oA = nb * 16 + col, oB = oA + 16;
  float bA = b_out[oA], bB = b_out[oB];
  f4 lo, hi;
#pragma unroll
  for (int e = 0; e < 4; ++e) { lo[e] = c0[e] + bA; hi[e] = c0[4 + e] + bA; }
  float* pA = out + (size_t)(b * 64 + oA) * SPAT + s0;
  *(f4*)pA = lo; *(f4*)(pA + 4) = hi;
#pragma unroll
  for (int e = 0; e < 4; ++e) { lo[e] = c1[e] + bB; hi[e] = c1[4 + e] + bB; }
  float* pB = out + (size_t)(b * 64 + oB) * SPAT + s0;
  *(f4*)pB = lo; *(f4*)(pB + 4) = hi;
}

// ---------- host launcher ----------
extern "C" void kernel_launch(void* const* d_in, const int* in_sizes, int n_in,
                              void* d_out, int out_size, void* d_ws, size_t ws_size,
                              hipStream_t stream) {
  const float* x     = (const float*)d_in[0];
  const float* gamma = (const float*)d_in[1];
  const float* beta  = (const float*)d_in[2];
  const float* w_qkv = (const float*)d_in[3];
  const float* w_out = (const float*)d_in[4];
  const float* b_out = (const float*)d_in[5];
  float* out = (float*)d_out;

  char* ws = (char*)d_ws;
  size_t off = 0;
  auto alloc = [&](size_t bytes) -> void* {
    void* p = ws + off;
    off += (bytes + 255) & ~(size_t)255;
    return p;
  };
  _Float16* xn   = (_Float16*)alloc((size_t)NVOX * CIN * 2);
  _Float16* wqh  = (_Float16*)alloc((size_t)OC3 * CIN * 2);
  _Float16* woh  = (_Float16*)alloc((size_t)DH * INNER * 2);
  _Float16* qn   = (_Float16*)alloc((size_t)BH * SPAT * DH * 2);  // aliased as attn_out
  _Float16* kn   = (_Float16*)alloc((size_t)BH * SPAT * DH * 2);
  _Float16* vn   = (_Float16*)alloc((size_t)BH * SPAT * DH * 2);
  float*    probe= (float*)alloc((size_t)BH * 64 * 4);
  float*    ksd  = (float*)alloc((size_t)BH * 64 * 32 * 4);
  float*    ksh  = (float*)alloc((size_t)BH * 64 * 32 * 4);
  float*    ksw  = (float*)alloc((size_t)BH * 64 * 32 * 4);
  int*      idx  = (int*)alloc((size_t)BH * 15 * 4);
  _Float16* kf   = (_Float16*)alloc((size_t)BH * LPAD * DH * 2);
  _Float16* vfT  = (_Float16*)alloc((size_t)BH * DH * LPAD * 2);
  _Float16* attn_out = qn;

  k_cvt<<<768, 256, 0, stream>>>(w_qkv, w_out, wqh, woh);
  k_layernorm<<<NVOX / 256, 256, 0, stream>>>(x, gamma, beta, xn);
  k_qkv<<<dim3(24, 1024), 256, 0, stream>>>(xn, wqh, qn, kn, vn);
  k_reduce<<<BH, 256, 0, stream>>>(qn, kn, probe, ksd, ksh, ksw);
  k_topk<<<BH, 32, 0, stream>>>(probe, ksd, ksh, ksw, idx);
  k_gather<<<BH, 256, 0, stream>>>(kn, vn, idx, kf, vfT);
  k_attn<<<dim3(512, BH), 256, 0, stream>>>(qn, kf, vfT, attn_out);
  k_out<<<1024, 256, 0, stream>>>(attn_out, woh, b_out, out);
}
